// GraphEvolution_88038239633979
// MI455X (gfx1250) — matrix operand order repaired and checked by reading, not compiled, not run on hardware
//
#include <hip/hip_runtime.h>
#include <hip/hip_fp16.h>

typedef __attribute__((ext_vector_type(16))) _Float16 v16h;
typedef __attribute__((ext_vector_type(8)))  _Float16 v8h;
typedef __attribute__((ext_vector_type(8)))  float    v8f;

#define DEVINL __device__ __forceinline__

// ---------------- WMMA helpers (CDNA5 v_wmma_f32_16x16x32_f16) ----------------

DEVINL v8f wmma_f16f32(v16h a, v16h b, v8f c) {
  // 8 args: (neg_a, A, neg_b, B, c_mod, C, reuse_a, reuse_b)
  return __builtin_amdgcn_wmma_f32_16x16x32_f16(false, a, false, b, (short)0, c,
                                                false, false);
}

// A-matrix fragment (16x32 f16) from an LDS tile A[16][stride].
// ISA layout: lanes 0-15 row M=lane, K=k0..k0+7 and k0+16..k0+23;
//             lanes 16-31 row M=lane-16, K=k0+8..k0+15 and k0+24..k0+31.
DEVINL v16h load_a_frag(const _Float16* A, int stride, int k0, int lane) {
  int row = lane & 15;
  int kb  = k0 + ((lane >> 4) << 3);
  const _Float16* p = A + row * stride + kb;
  v8h lo = *(const v8h*)(p);
  v8h hi = *(const v8h*)(p + 16);
  v16h r;
#pragma unroll
  for (int i = 0; i < 8; ++i) { r[i] = lo[i]; r[i + 8] = hi[i]; }
  return r;
}

// B-matrix fragment: pre-packed contiguous per (ntile, kstep, lane): 16 halfs.
DEVINL v16h load_b_frag(const _Float16* Bp, int nt, int ks, int ksteps, int lane) {
  return *(const v16h*)(Bp + ((size_t)(nt * ksteps + ks) * 32 + lane) * 16);
}

// GEMM of one 16-row tile: T[16][tstride] (+bias) = A(16xK,f16) * B(KxN,f16).
// ntiles split across the block's 4 waves.
DEVINL void gemm_lds(const _Float16* A, int astride, const _Float16* Bp,
                     const float* bias, float* T, int tstride,
                     int ksteps, int ntiles, int wave, int lane) {
  for (int nt = wave; nt < ntiles; nt += 4) {
    v8f acc = {};
    for (int ks = 0; ks < ksteps; ++ks)
      acc = wmma_f16f32(load_a_frag(A, astride, ks * 32, lane),
                        load_b_frag(Bp, nt, ks, ksteps, lane), acc);
    int col = nt * 16 + (lane & 15);
    int r0  = (lane >> 4) << 3;
    float bv = bias ? bias[col] : 0.0f;
#pragma unroll
    for (int j = 0; j < 8; ++j) T[(r0 + j) * tstride + col] = acc[j] + bv;
  }
}

// ---------------- Weight pre-pack into WMMA B layout (f16, K padded) ----------

__global__ void pack_b_kernel(const float* __restrict__ W, _Float16* __restrict__ out,
                              int K, int N) {
  int ksteps = (K + 31) >> 5;
  int ntiles = N >> 4;
  int idx = blockIdx.x * blockDim.x + threadIdx.x;
  int total = ntiles * ksteps * 32;
  if (idx >= total) return;
  int lane = idx & 31;
  int ks   = (idx >> 5) % ksteps;
  int nt   = idx / (32 * ksteps);
  int col  = nt * 16 + (lane & 15);
  int kb   = ks * 32 + ((lane >> 4) << 3);  // lanes 0-15: K 0..7,16..23; 16-31: K 8..15,24..31
  _Float16* o = out + (size_t)idx * 16;
#pragma unroll
  for (int e = 0; e < 16; ++e) {
    int k = kb + e + ((e >= 8) ? 8 : 0);
    o[e] = (_Float16)((k < K) ? W[(size_t)k * N + col] : 0.0f);
  }
}

// selfee[128] = (-1 vector) @ We   (GATv2 self-loop edge embedding)
__global__ void prep_selfee_kernel(const float* __restrict__ We, float* selfee) {
  int c = threadIdx.x;  // 128
  float s = 0.f;
#pragma unroll
  for (int m = 0; m < 6; ++m) s -= We[m * 128 + c];
  selfee[c] = s;
}

// ---------------- Node projections xl = x@Wl+bl, xr = x@Wr+br (WMMA) ----------

__global__ void proj_kernel(const float* __restrict__ x,
                            const _Float16* __restrict__ pWl, const float* __restrict__ bl,
                            const _Float16* __restrict__ pWr, const float* __restrict__ br,
                            float* __restrict__ xl, float* __restrict__ xr, int Nn) {
  __shared__ __align__(32) _Float16 A16[16 * 64];
  int tid = threadIdx.x, wave = tid >> 5, lane = tid & 31;
  int row0 = blockIdx.x * 16;
  bool full = (row0 + 16 <= Nn);
  if (full) {
    // vectorized b128 loads: 16x64 = 256 float4
    const float4* x4 = (const float4*)(x + (size_t)row0 * 64);
    for (int i = tid; i < 256; i += 128) {
      float4 v = x4[i];
      int b = i * 4;
      A16[b + 0] = (_Float16)v.x; A16[b + 1] = (_Float16)v.y;
      A16[b + 2] = (_Float16)v.z; A16[b + 3] = (_Float16)v.w;
    }
  } else {
    for (int i = tid; i < 16 * 64; i += 128) {
      int r = i >> 6, c = i & 63;
      int rr = row0 + r; if (rr >= Nn) rr = Nn - 1;
      A16[i] = (_Float16)x[(size_t)rr * 64 + c];
    }
  }
  __syncthreads();
  for (int nt = wave; nt < 16; nt += 4) {
    const _Float16* Bp  = (nt < 8) ? pWl : pWr;
    const float* bias   = (nt < 8) ? bl : br;
    float* dstp         = (nt < 8) ? xl : xr;
    int ntl = nt & 7;
    v8f acc = {};
    for (int ks = 0; ks < 2; ++ks)
      acc = wmma_f16f32(load_a_frag(A16, 64, ks * 32, lane),
                        load_b_frag(Bp, ntl, ks, 2, lane), acc);
    int col = ntl * 16 + (lane & 15);
    int r0  = (lane >> 4) << 3;
    float bv = bias[col];
    if (full) {
      float* o = dstp + (size_t)(row0 + r0) * 128 + col;
#pragma unroll
      for (int j = 0; j < 8; ++j) o[(size_t)j * 128] = acc[j] + bv;
    } else {
#pragma unroll
      for (int j = 0; j < 8; ++j) {
        int rr = row0 + r0 + j;
        if (rr < Nn) dstp[(size_t)rr * 128 + col] = acc[j] + bv;
      }
    }
  }
}

// ---------------- Edge encoder + GATv2 softmax passes -----------------------

struct EncLayerP {
  const float *Wv, *bv, *Wo, *bo, *W1, *b1f, *W2, *b2f, *g1, *b1, *g2, *b2;
};
struct EdgeP {
  EncLayerP enc[2];
  const float *rzW, *rzb, *We, *att;
  const int *src, *dst;
  const float *eattr, *xl, *xr;
  float *exbuf, *den;
  int E;
};

DEVINL void run_encoder(const EdgeP& p, const float* __restrict__ ein, float* ea6) {
  float e[8];
  {
    const float4* e4 = (const float4*)ein;
    float4 v0 = e4[0], v1 = e4[1];
    e[0] = v0.x; e[1] = v0.y; e[2] = v0.z; e[3] = v0.w;
    e[4] = v1.x; e[5] = v1.y; e[6] = v1.z; e[7] = v1.w;
  }
#pragma unroll
  for (int l = 0; l < 2; ++l) {
    const EncLayerP& L = p.enc[l];
    float t[8], a[8], r[8];
#pragma unroll
    for (int i = 0; i < 8; ++i) {
      float s = L.bv[i];
#pragma unroll
      for (int j = 0; j < 8; ++j) s += e[j] * L.Wv[j * 8 + i];
      t[i] = s;
    }
#pragma unroll
    for (int i = 0; i < 8; ++i) {
      float s = L.bo[i];
#pragma unroll
      for (int j = 0; j < 8; ++j) s += t[j] * L.Wo[j * 8 + i];
      a[i] = s;
    }
    float m = 0.f;
#pragma unroll
    for (int i = 0; i < 8; ++i) { r[i] = e[i] + a[i]; m += r[i]; }
    m *= 0.125f;
    float var = 0.f;
#pragma unroll
    for (int i = 0; i < 8; ++i) { float d = r[i] - m; var += d * d; }
    float rs = rsqrtf(var * 0.125f + 1e-5f);
#pragma unroll
    for (int i = 0; i < 8; ++i) e[i] = (r[i] - m) * rs * L.g1[i] + L.b1[i];
    float h[16];
#pragma unroll
    for (int k = 0; k < 16; ++k) {
      float s = L.b1f[k];
#pragma unroll
      for (int j = 0; j < 8; ++j) s += e[j] * L.W1[j * 16 + k];
      h[k] = (s > 0.f) ? s : (expf(s) - 1.0f);  // elu
    }
    float f[8];
#pragma unroll
    for (int i = 0; i < 8; ++i) {
      float s = L.b2f[i];
#pragma unroll
      for (int k = 0; k < 16; ++k) s += h[k] * L.W2[k * 8 + i];
      f[i] = s;
    }
    m = 0.f;
#pragma unroll
    for (int i = 0; i < 8; ++i) { r[i] = e[i] + f[i]; m += r[i]; }
    m *= 0.125f;
    var = 0.f;
#pragma unroll
    for (int i = 0; i < 8; ++i) { float d = r[i] - m; var += d * d; }
    rs = rsqrtf(var * 0.125f + 1e-5f);
#pragma unroll
    for (int i = 0; i < 8; ++i) e[i] = (r[i] - m) * rs * L.g2[i] + L.b2[i];
  }
#pragma unroll
  for (int mo = 0; mo < 6; ++mo) {
    float s = p.rzb[mo];
#pragma unroll
    for (int j = 0; j < 8; ++j) s += e[j] * p.rzW[j * 6 + mo];
    ea6[mo] = (s > 0.f) ? s : (expf(s) - 1.0f);
  }
}

__global__ void edge_pass1_kernel(EdgeP p) {
  int e = blockIdx.x * 256 + threadIdx.x;
  if (e >= p.E) return;
  int s = p.src[e], d = p.dst[e];
  __builtin_prefetch(p.xl + (size_t)s * 128, 0, 0);   // global_prefetch_b8
  __builtin_prefetch(p.xr + (size_t)d * 128, 0, 0);
  float ea6[6];
  run_encoder(p, p.eattr + (size_t)e * 8, ea6);
  const float* xls = p.xl + (size_t)s * 128;
  const float* xrd = p.xr + (size_t)d * 128;
#pragma unroll
  for (int h = 0; h < 8; ++h) {
    float lg = 0.f;
#pragma unroll
    for (int c = 0; c < 16; ++c) {
      int ch = h * 16 + c;
      float eec = 0.f;
#pragma unroll
      for (int m = 0; m < 6; ++m) eec += ea6[m] * p.We[m * 128 + ch];
      float v = xls[ch] + xrd[ch] + eec;
      v = (v > 0.f) ? v : 0.2f * v;  // leaky_relu(0.2)
      lg += v * p.att[ch];
    }
    float ex = expf(lg);
    p.exbuf[(size_t)e * 8 + h] = ex;
    atomicAdd(&p.den[(size_t)d * 8 + h], ex);
  }
}

// Self loops: initialize den (no memset needed) and stash ex_self.
__global__ void self_pass1_kernel(const float* __restrict__ xl, const float* __restrict__ xr,
                                  const float* __restrict__ selfee, const float* __restrict__ att,
                                  float* __restrict__ den, float* __restrict__ exself, int Nn) {
  int i = blockIdx.x * 256 + threadIdx.x;
  if (i >= Nn) return;
  const float* a = xl + (size_t)i * 128;
  const float* b = xr + (size_t)i * 128;
#pragma unroll
  for (int h = 0; h < 8; ++h) {
    float lg = 0.f;
#pragma unroll
    for (int c = 0; c < 16; ++c) {
      int ch = h * 16 + c;
      float v = a[ch] + b[ch] + selfee[ch];
      v = (v > 0.f) ? v : 0.2f * v;
      lg += v * att[ch];
    }
    float ex = expf(lg);
    den[(size_t)i * 8 + h]    = ex;
    exself[(size_t)i * 8 + h] = ex;
  }
}

// y[i] = xl[i]*alpha_self + gat_bias   (then edge_pass2 accumulates on top)
__global__ void node_init_kernel(const float* __restrict__ xl, const float* __restrict__ exself,
                                 const float* __restrict__ den, const float* __restrict__ gbias,
                                 float* __restrict__ out, int Nn) {
  int i = blockIdx.x * 256 + threadIdx.x;
  if (i >= Nn) return;
  float al[8];
#pragma unroll
  for (int h = 0; h < 8; ++h)
    al[h] = exself[(size_t)i * 8 + h] / den[(size_t)i * 8 + h];
  const float* a = xl + (size_t)i * 128;
  float* o = out + (size_t)i * 128;
#pragma unroll 8
  for (int ch = 0; ch < 128; ++ch) o[ch] = a[ch] * al[ch >> 4] + gbias[ch];
}

__global__ void edge_pass2_kernel(const int* __restrict__ src, const int* __restrict__ dst,
                                  const float* __restrict__ xl, const float* __restrict__ exbuf,
                                  const float* __restrict__ den, float* __restrict__ out, int E) {
  int e = blockIdx.x * 256 + threadIdx.x;
  if (e >= E) return;
  int s = src[e], d = dst[e];
  __builtin_prefetch(xl + (size_t)s * 128, 0, 0);
  float al[8];
#pragma unroll
  for (int h = 0; h < 8; ++h)
    al[h] = exbuf[(size_t)e * 8 + h] / den[(size_t)d * 8 + h];
  const float* a = xl + (size_t)s * 128;
  float* o = out + (size_t)d * 128;
#pragma unroll 4
  for (int ch = 0; ch < 128; ++ch)
    atomicAdd(&o[ch], a[ch] * al[ch >> 4]);  // global_atomic_add_f32
}

// ---------------- Fused 2-layer decoder + final projection (WMMA) ------------

struct DecLayerP {
  const _Float16 *Wv, *Wo, *cWv, *cWo, *W1, *W2;
  const float *bv, *bo, *cbv, *cbo, *b1, *b2, *g1, *bn1, *g2, *bn2, *g3, *bn3;
};
struct DecP {
  DecLayerP L[2];
  const _Float16* Wrz;
  const float *brz, *x, *yin;
  float* out;
  int Nn;
};

DEVINL void ln_res(float* Y, const float* T, const float* g, const float* b, int tid) {
  if (tid < 16) {
    const float* yr = Y + tid * 128;
    const float* tr = T + tid * 128;
    float m = 0.f;
    for (int c = 0; c < 128; ++c) m += yr[c] + tr[c];
    m *= (1.0f / 128.0f);
    float var = 0.f;
    for (int c = 0; c < 128; ++c) { float d = yr[c] + tr[c] - m; var += d * d; }
    float rs = rsqrtf(var * (1.0f / 128.0f) + 1e-5f);
    float* yw = Y + tid * 128;
    for (int c = 0; c < 128; ++c) {
      float d = yr[c] + tr[c];
      yw[c] = (d - m) * rs * g[c] + b[c];
    }
  }
}

__global__ void decoder_kernel(DecP p) {
  __shared__ __align__(16) float    Y[16 * 128];
  __shared__ __align__(16) float    T[16 * 128];
  __shared__ __align__(32) _Float16 A16[16 * 128];
  __shared__ __align__(32) _Float16 M16[16 * 128];
  __shared__ __align__(32) _Float16 F16[16 * 32];
  int tid = threadIdx.x, wave = tid >> 5, lane = tid & 31;
  int row0 = blockIdx.x * 16;
  bool full = (row0 + 16 <= p.Nn);

  if (full) {
    // vectorized b128 loads: 16x128 = 512 float4
    const float4* y4 = (const float4*)(p.yin + (size_t)row0 * 128);
    for (int i = tid; i < 512; i += 128) {
      float4 v = y4[i];
      int b = i * 4;
      Y[b + 0] = v.x; Y[b + 1] = v.y; Y[b + 2] = v.z; Y[b + 3] = v.w;
      M16[b + 0] = (_Float16)v.x; M16[b + 1] = (_Float16)v.y;
      M16[b + 2] = (_Float16)v.z; M16[b + 3] = (_Float16)v.w;
    }
  } else {
    for (int i = tid; i < 2048; i += 128) {
      int r = i >> 7, c = i & 127;
      int rr = row0 + r; if (rr >= p.Nn) rr = p.Nn - 1;
      float v = p.yin[(size_t)rr * 128 + c];
      Y[i] = v; M16[i] = (_Float16)v;   // mem = y (cross-attn source, fixed)
    }
  }
  __syncthreads();

#pragma unroll
  for (int l = 0; l < 2; ++l) {
    DecLayerP L = p.L[l];
    // ---- self "attention" (seq len 1): sa = (y Wv + bv) Wo + bo
    for (int i = tid; i < 2048; i += 128) A16[i] = (_Float16)Y[i];
    __syncthreads();
    gemm_lds(A16, 128, L.Wv, L.bv, T, 128, 4, 8, wave, lane);
    __syncthreads();
    for (int i = tid; i < 2048; i += 128) A16[i] = (_Float16)T[i];
    __syncthreads();
    gemm_lds(A16, 128, L.Wo, L.bo, T, 128, 4, 8, wave, lane);
    __syncthreads();
    ln_res(Y, T, L.g1, L.bn1, tid);
    __syncthreads();
    // ---- cross "attention": ca = (mem cWv + cbv) cWo + cbo
    gemm_lds(M16, 128, L.cWv, L.cbv, T, 128, 4, 8, wave, lane);
    __syncthreads();
    for (int i = tid; i < 2048; i += 128) A16[i] = (_Float16)T[i];
    __syncthreads();
    gemm_lds(A16, 128, L.cWo, L.cbo, T, 128, 4, 8, wave, lane);
    __syncthreads();
    ln_res(Y, T, L.g2, L.bn2, tid);
    __syncthreads();
    // ---- FFN: f = elu(y W1 + b1) W2 + b2  (K of W2 padded 16 -> 32)
    for (int i = tid; i < 2048; i += 128) A16[i] = (_Float16)Y[i];
    __syncthreads();
    gemm_lds(A16, 128, L.W1, L.b1, T, 128, 4, 1, wave, lane);
    __syncthreads();
    for (int i = tid; i < 512; i += 128) {
      int r = i >> 5, c = i & 31;
      float v = 0.f;
      if (c < 16) { float t = T[r * 128 + c]; v = (t > 0.f) ? t : (expf(t) - 1.0f); }
      F16[i] = (_Float16)v;
    }
    __syncthreads();
    gemm_lds(F16, 32, L.W2, L.b2, T, 128, 1, 8, wave, lane);
    __syncthreads();
    ln_res(Y, T, L.g3, L.bn3, tid);
    __syncthreads();
  }

  // ---- final: out = y @ dec_rz_W + dec_rz_b + x
  for (int i = tid; i < 2048; i += 128) A16[i] = (_Float16)Y[i];
  __syncthreads();
  for (int nt = wave; nt < 4; nt += 4) {
    v8f acc = {};
    for (int ks = 0; ks < 4; ++ks)
      acc = wmma_f16f32(load_a_frag(A16, 128, ks * 32, lane),
                        load_b_frag(p.Wrz, nt, ks, 4, lane), acc);
    int col = nt * 16 + (lane & 15);
    int r0  = (lane >> 4) << 3;
    float bb = p.brz[col];
    if (full) {
      float*       o  = p.out + (size_t)(row0 + r0) * 64 + col;
      const float* xp = p.x   + (size_t)(row0 + r0) * 64 + col;
#pragma unroll
      for (int j = 0; j < 8; ++j) o[(size_t)j * 64] = acc[j] + bb + xp[(size_t)j * 64];
    } else {
#pragma unroll
      for (int j = 0; j < 8; ++j) {
        int rr = row0 + r0 + j;
        if (rr < p.Nn)
          p.out[(size_t)rr * 64 + col] = acc[j] + bb + p.x[(size_t)rr * 64 + col];
      }
    }
  }
}

// ---------------- Host-side orchestration ------------------------------------

extern "C" void kernel_launch(void* const* d_in, const int* in_sizes, int n_in,
                              void* d_out, int out_size, void* d_ws, size_t ws_size,
                              hipStream_t stream) {
  (void)n_in; (void)out_size; (void)ws_size;
  const int C = 64, ED = 8;
  const int Nn = in_sizes[0] / C;   // B*N = 50000
  const int E  = in_sizes[2] / ED;  // 1,000,000

  const float* x     = (const float*)d_in[0];
  const int*   eidx  = (const int*)d_in[1];
  const float* eattr = (const float*)d_in[2];

  // d_in layout (setup_inputs dict insertion order, params flattened):
  const int I_ENC0 = 3, I_ENC1 = 15, I_DEC0 = 27, I_DEC1 = 45;
  const int I_RZW = 63, I_RZB = 64, I_WL = 65, I_BL = 66, I_WR = 67, I_BR = 68;
  const int I_WE = 69, I_ATT = 70, I_GB = 71, I_DRZW = 72, I_DRZB = 73;
  auto F = [&](int i) -> const float* { return (const float*)d_in[i]; };

  // workspace layout
  char* ws = (char*)d_ws;
  size_t off = 0;
  auto alloc = [&](size_t bytes) -> char* {
    char* p = ws + off;
    off = (off + bytes + 255) & ~(size_t)255;
    return p;
  };
  float* xl     = (float*)alloc((size_t)Nn * 128 * 4);
  float* xr     = (float*)alloc((size_t)Nn * 128 * 4);
  float* yb     = (float*)alloc((size_t)Nn * 128 * 4);
  float* exbuf  = (float*)alloc((size_t)E * 8 * 4);
  float* den    = (float*)alloc((size_t)Nn * 8 * 4);
  float* exself = (float*)alloc((size_t)Nn * 8 * 4);
  float* selfee = (float*)alloc(128 * 4);
  _Float16* pk  = (_Float16*)alloc((size_t)167936 * sizeof(_Float16));

  // packed weight sub-offsets (in halfs; all 32B aligned)
  _Float16* pWl = pk;
  _Float16* pWr = pk + 8192;
  _Float16* pD[2][6];
  size_t po = 16384;
  for (int l = 0; l < 2; ++l) {
    for (int m = 0; m < 4; ++m) { pD[l][m] = pk + po; po += 16384; } // 128x128
    pD[l][4] = pk + po; po += 2048;   // W1 128x16
    pD[l][5] = pk + po; po += 4096;   // W2 16x128 (K padded to 32)
  }
  _Float16* pRz = pk + po;            // 128x64

  auto pack = [&](const float* W, _Float16* out, int K, int N) {
    int tot = (N / 16) * ((K + 31) / 32) * 32;
    pack_b_kernel<<<(tot + 127) / 128, 128, 0, stream>>>(W, out, K, N);
  };
  pack(F(I_WL), pWl, 64, 128);
  pack(F(I_WR), pWr, 64, 128);
  for (int l = 0; l < 2; ++l) {
    int base = l ? I_DEC1 : I_DEC0;
    pack(F(base + 0),  pD[l][0], 128, 128);  // sWv
    pack(F(base + 2),  pD[l][1], 128, 128);  // sWo
    pack(F(base + 4),  pD[l][2], 128, 128);  // cWv
    pack(F(base + 6),  pD[l][3], 128, 128);  // cWo
    pack(F(base + 8),  pD[l][4], 128, 16);   // W1
    pack(F(base + 10), pD[l][5], 16, 128);   // W2 (K padded)
  }
  pack(F(I_DRZW), pRz, 128, 64);
  prep_selfee_kernel<<<1, 128, 0, stream>>>(F(I_WE), selfee);

  // 1) node projections (WMMA)
  proj_kernel<<<(Nn + 15) / 16, 128, 0, stream>>>(x, pWl, F(I_BL), pWr, F(I_BR),
                                                  xl, xr, Nn);
  // 2) softmax denominators: self loops init den, edges atomically accumulate
  self_pass1_kernel<<<(Nn + 255) / 256, 256, 0, stream>>>(xl, xr, selfee, F(I_ATT),
                                                          den, exself, Nn);
  EdgeP ep;
  for (int l = 0; l < 2; ++l) {
    int b = l ? I_ENC1 : I_ENC0;
    ep.enc[l] = {F(b + 0), F(b + 1), F(b + 2), F(b + 3), F(b + 4), F(b + 5),
                 F(b + 6), F(b + 7), F(b + 8), F(b + 9), F(b + 10), F(b + 11)};
  }
  ep.rzW = F(I_RZW); ep.rzb = F(I_RZB); ep.We = F(I_WE); ep.att = F(I_ATT);
  ep.src = eidx; ep.dst = eidx + E;
  ep.eattr = eattr; ep.xl = xl; ep.xr = xr;
  ep.exbuf = exbuf; ep.den = den; ep.E = E;
  edge_pass1_kernel<<<(E + 255) / 256, 256, 0, stream>>>(ep);

  // 3) weighted aggregation: y = sum alpha * xl[src] (+ self) + gat_bias
  node_init_kernel<<<(Nn + 255) / 256, 256, 0, stream>>>(xl, exself, den, F(I_GB),
                                                         yb, Nn);
  edge_pass2_kernel<<<(E + 255) / 256, 256, 0, stream>>>(eidx, eidx + E, xl, exbuf,
                                                         den, yb, E);

  // 4) fused decoder + final projection + residual (WMMA)
  DecP dp;
  for (int l = 0; l < 2; ++l) {
    int b = l ? I_DEC1 : I_DEC0;
    dp.L[l].Wv = pD[l][0]; dp.L[l].Wo = pD[l][1];
    dp.L[l].cWv = pD[l][2]; dp.L[l].cWo = pD[l][3];
    dp.L[l].W1 = pD[l][4]; dp.L[l].W2 = pD[l][5];
    dp.L[l].bv = F(b + 1); dp.L[l].bo = F(b + 3);
    dp.L[l].cbv = F(b + 5); dp.L[l].cbo = F(b + 7);
    dp.L[l].b1 = F(b + 9); dp.L[l].b2 = F(b + 11);
    dp.L[l].g1 = F(b + 12); dp.L[l].bn1 = F(b + 13);
    dp.L[l].g2 = F(b + 14); dp.L[l].bn2 = F(b + 15);
    dp.L[l].g3 = F(b + 16); dp.L[l].bn3 = F(b + 17);
  }
  dp.Wrz = pRz; dp.brz = F(I_DRZB);
  dp.x = x; dp.yin = yb; dp.out = (float*)d_out; dp.Nn = Nn;
  decoder_kernel<<<(Nn + 15) / 16, 128, 0, stream>>>(dp);
}
